// SDSG4_3496103379545
// MI455X (gfx1250) — matrix operand order repaired and checked by reading, not compiled, not run on hardware
//
#include <hip/hip_runtime.h>
#include <hip/hip_fp16.h>

typedef __attribute__((ext_vector_type(16))) _Float16 v16h;
typedef __attribute__((ext_vector_type(8)))  float    v8f;

// K offset within A fragment (16-bit A 16x32 layout, ISA 7.12.2)
__device__ __forceinline__ int koffA(int j, int hi) {
    return (j < 4) ? (8 * hi + 2 * j) : (16 + 8 * hi + 2 * (j - 4));
}

__device__ __forceinline__ void wave_minmax(float v, float& mn, float& mx) {
    mn = v; mx = v;
#pragma unroll
    for (int d = 16; d > 0; d >>= 1) {
        mn = fminf(mn, __shfl_xor(mn, d, 32));
        mx = fmaxf(mx, __shfl_xor(mx, d, 32));
    }
}

// ---------------- degree / dinv ----------------
__global__ void k_init_deg(float* deg, int n) {
    int i = blockIdx.x * 256 + threadIdx.x;
    if (i < n) deg[i] = 1.0f;          // self-loop contributes 1
}

__global__ void k_count_deg(const int* __restrict__ col, float* deg, int e) {
    int i = blockIdx.x * 256 + threadIdx.x;
    if (i < e) unsafeAtomicAdd(&deg[col[i]], 1.0f);
}

__global__ void k_dinv(float* deg, int n) {
    int i = blockIdx.x * 256 + threadIdx.x;
    if (i < n) {
        float d = deg[i];
        deg[i] = (d > 0.0f) ? rsqrtf(d) : 0.0f;
    }
}

// ---------------- fc1: x[N,128] @ W[32,128]^T + b, relu ----------------
__global__ void k_fc1(const float* __restrict__ x, const float* __restrict__ W,
                      const float* __restrict__ bias, float* __restrict__ out, int n) {
    int lane = threadIdx.x & 31, wv = threadIdx.x >> 5;
    long i0 = ((long)blockIdx.x * 4 + wv) * 16;   // 16-node tile per wave
    if (i0 >= n) return;
    int m = lane & 15, hi = lane >> 4;
    const float* arow = x + (i0 + m) * 128;
    v8f c0 = {}, c1 = {};
    for (int kk = 0; kk < 4; ++kk) {
        int kbase = kk * 32;
        v16h a, b0, b1;
#pragma unroll
        for (int j = 0; j < 8; ++j) {
            int k = kbase + koffA(j, hi);
            a[2 * j]     = (_Float16)arow[k];
            a[2 * j + 1] = (_Float16)arow[k + 1];
        }
#pragma unroll
        for (int j = 0; j < 8; ++j) {
            int k = kbase + koffA(j, hi);          // B 32x16 layout
            b0[2 * j]     = (_Float16)W[(long)m * 128 + k];
            b0[2 * j + 1] = (_Float16)W[(long)m * 128 + k + 1];
            b1[2 * j]     = (_Float16)W[(long)(16 + m) * 128 + k];
            b1[2 * j + 1] = (_Float16)W[(long)(16 + m) * 128 + k + 1];
        }
        c0 = __builtin_amdgcn_wmma_f32_16x16x32_f16(false, a, false, b0, (short)0, c0, false, false);
        c1 = __builtin_amdgcn_wmma_f32_16x16x32_f16(false, a, false, b1, (short)0, c1, false, false);
    }
#pragma unroll
    for (int v = 0; v < 8; ++v) {
        long r = i0 + v + 8 * hi;                  // C/D row = v + 8*hi
        out[r * 32 + m]      = fmaxf(c0[v] + bias[m], 0.0f);
        out[r * 32 + 16 + m] = fmaxf(c1[v] + bias[16 + m], 0.0f);
    }
}

// ---------------- per-row min-max norm in place, 32 feats ----------------
__global__ void k_rownorm(float* x, int n) {
    int lane = threadIdx.x & 31, wv = threadIdx.x >> 5;
    long i = (long)blockIdx.x * 8 + wv;
    if (i >= n) return;
    float v = x[i * 32 + lane], mn, mx;
    wave_minmax(v, mn, mx);
    x[i * 32 + lane] = 2.0f * (v - mn) / (mx - mn + 1e-8f) - 1.0f;
}

// ---------------- self-loop init: agg = dinv^2 * xin ----------------
__global__ void k_selfinit(const float* __restrict__ xin, const float* __restrict__ dinv,
                           float* __restrict__ agg, int n) {
    long t = (long)blockIdx.x * 256 + threadIdx.x;   // n*32 threads
    if (t >= (long)n * 32) return;
    int i = (int)(t >> 5);
    float s = dinv[i];
    agg[t] = s * s * xin[t];
}

// ---------------- edge scatter: agg[col] += dinv[row]*dinv[col]*xin[row] ----------------
__global__ void k_edge(const int* __restrict__ ei, const float* __restrict__ dinv,
                       const float* __restrict__ xin, float* __restrict__ agg, int e) {
    long t = (long)blockIdx.x * 256 + threadIdx.x;   // e*8 threads, float4 each
    if (t >= (long)e * 8) return;
    int eid = (int)(t >> 3), q = (int)(t & 7);
    int r = ei[eid], c = ei[(long)e + eid];
    float w = dinv[r] * dinv[c];
    float4 v = *(const float4*)(xin + (long)r * 32 + q * 4);
    float* dst = agg + (long)c * 32 + q * 4;
    unsafeAtomicAdd(dst + 0, w * v.x);
    unsafeAtomicAdd(dst + 1, w * v.y);
    unsafeAtomicAdd(dst + 2, w * v.z);
    unsafeAtomicAdd(dst + 3, w * v.w);
}

// ---------------- 32x32 linear: out = A[N,32] @ W[32,32]^T + b ----------------
__global__ void k_gemm32(const float* __restrict__ A, const float* __restrict__ W,
                         const float* __restrict__ bias, float* __restrict__ out, int n) {
    int lane = threadIdx.x & 31, wv = threadIdx.x >> 5;
    long i0 = ((long)blockIdx.x * 4 + wv) * 16;
    if (i0 >= n) return;
    int m = lane & 15, hi = lane >> 4;
    const float* arow = A + (i0 + m) * 32;
    v16h a, b0, b1;
#pragma unroll
    for (int j = 0; j < 8; ++j) {
        int k = koffA(j, hi);
        a[2 * j]     = (_Float16)arow[k];
        a[2 * j + 1] = (_Float16)arow[k + 1];
    }
#pragma unroll
    for (int j = 0; j < 8; ++j) {
        int k = koffA(j, hi);
        b0[2 * j]     = (_Float16)W[m * 32 + k];
        b0[2 * j + 1] = (_Float16)W[m * 32 + k + 1];
        b1[2 * j]     = (_Float16)W[(16 + m) * 32 + k];
        b1[2 * j + 1] = (_Float16)W[(16 + m) * 32 + k + 1];
    }
    v8f c0 = {}, c1 = {};
    c0 = __builtin_amdgcn_wmma_f32_16x16x32_f16(false, a, false, b0, (short)0, c0, false, false);
    c1 = __builtin_amdgcn_wmma_f32_16x16x32_f16(false, a, false, b1, (short)0, c1, false, false);
#pragma unroll
    for (int v = 0; v < 8; ++v) {
        long r = i0 + v + 8 * hi;
        out[r * 32 + m]      = c0[v] + bias[m];
        out[r * 32 + 16 + m] = c1[v] + bias[16 + m];
    }
}

// ---------------- finalize: build xx4 in LDS, WMMA with W_fc4[64,128] ----------------
__global__ void k_final(const float* __restrict__ x0, const float* __restrict__ x1,
                        const float* __restrict__ x2, const float* __restrict__ x3,
                        const float* __restrict__ W, const float* __restrict__ bias,
                        float* __restrict__ out, int n) {
    __shared__ float xx[16 * 128];                 // 8 KB tile of xx4
    int lane = threadIdx.x & 31, wv = threadIdx.x >> 5;
    long i0 = (long)blockIdx.x * 16;
    if (i0 >= n) return;
    // stage 1: each wave builds 4 rows of xx4 (concat + mynorm diffs)
    for (int s = 0; s < 4; ++s) {
        int  loc  = wv * 4 + s;
        long node = i0 + loc;
        float a0 = x0[node * 32 + lane];
        float a1 = x1[node * 32 + lane];
        float a2 = x2[node * 32 + lane];
        float a3 = x3[node * 32 + lane];
        float mn0, mx0, mn1, mx1, mn2, mx2, mn3, mx3;
        wave_minmax(a0, mn0, mx0);
        wave_minmax(a1, mn1, mx1);
        wave_minmax(a2, mn2, mx2);
        wave_minmax(a3, mn3, mx3);
        float m0 = 2.0f * (a0 - mn0) / (mx0 - mn0 + 1e-8f) - 1.0f;
        float m1 = 2.0f * (a1 - mn1) / (mx1 - mn1 + 1e-8f) - 1.0f;
        float m2 = 2.0f * (a2 - mn2) / (mx2 - mn2 + 1e-8f) - 1.0f;
        float m3 = 2.0f * (a3 - mn3) / (mx3 - mn3 + 1e-8f) - 1.0f;
        xx[loc * 128 + lane]      = a0;        // x0
        xx[loc * 128 + 32 + lane] = a1;        // xx1 = x1
        xx[loc * 128 + 64 + lane] = m2 - m0;   // xx2
        xx[loc * 128 + 96 + lane] = m3 - m1;   // xx3
    }
    __syncthreads();
    // stage 2: wave wv computes output columns [wv*16, wv*16+16)
    int m = lane & 15, hi = lane >> 4;
    int o0 = wv * 16;
    v8f c = {};
    for (int kk = 0; kk < 4; ++kk) {
        int kbase = kk * 32;
        v16h a, bf;
#pragma unroll
        for (int j = 0; j < 8; ++j) {
            int k = kbase + koffA(j, hi);
            a[2 * j]     = (_Float16)xx[m * 128 + k];
            a[2 * j + 1] = (_Float16)xx[m * 128 + k + 1];
        }
#pragma unroll
        for (int j = 0; j < 8; ++j) {
            int k = kbase + koffA(j, hi);
            bf[2 * j]     = (_Float16)W[(long)(o0 + m) * 128 + k];
            bf[2 * j + 1] = (_Float16)W[(long)(o0 + m) * 128 + k + 1];
        }
        c = __builtin_amdgcn_wmma_f32_16x16x32_f16(false, a, false, bf, (short)0, c, false, false);
    }
#pragma unroll
    for (int v = 0; v < 8; ++v) {
        long r = i0 + v + 8 * hi;
        out[r * 64 + o0 + m] = c[v] + bias[o0 + m];
    }
}

extern "C" void kernel_launch(void* const* d_in, const int* in_sizes, int n_in,
                              void* d_out, int out_size, void* d_ws, size_t ws_size,
                              hipStream_t stream) {
    const float* x      = (const float*)d_in[0];
    const int*   ei     = (const int*)  d_in[1];
    const float* W_fc1  = (const float*)d_in[2];
    const float* b_fc1  = (const float*)d_in[3];
    const float* W_c1   = (const float*)d_in[4];
    const float* b_c1   = (const float*)d_in[5];
    const float* W_c2   = (const float*)d_in[6];
    const float* b_c2   = (const float*)d_in[7];
    const float* W_c3   = (const float*)d_in[8];
    const float* b_c3   = (const float*)d_in[9];
    const float* W_fc4  = (const float*)d_in[10];
    const float* b_fc4  = (const float*)d_in[11];
    float* out = (float*)d_out;

    const int N = in_sizes[0] / 128;
    const int E = in_sizes[1] / 2;

    float* dinv = (float*)d_ws;                    // N
    float* x0   = dinv + N;                        // N*32
    float* x1   = x0 + (size_t)N * 32;
    float* x2   = x1 + (size_t)N * 32;
    float* x3   = x2 + (size_t)N * 32;
    float* agg  = x3 + (size_t)N * 32;

    const int tiles   = (N + 15) / 16;
    const int gblocks = (tiles + 3) / 4;           // 4 waves/block, 16 nodes/wave
    const long nf     = (long)N * 32;
    const long et     = (long)E * 8;

    // gcn_norm
    k_init_deg<<<(N + 255) / 256, 256, 0, stream>>>(dinv, N);
    k_count_deg<<<(E + 255) / 256, 256, 0, stream>>>(ei + E, dinv, E);
    k_dinv<<<(N + 255) / 256, 256, 0, stream>>>(dinv, N);

    // x0 = mynorm(relu(x @ W_fc1^T + b))
    k_fc1<<<gblocks, 128, 0, stream>>>(x, W_fc1, b_fc1, x0, N);
    k_rownorm<<<(N + 7) / 8, 256, 0, stream>>>(x0, N);

    // conv1: x0 -> x1
    k_selfinit<<<(int)((nf + 255) / 256), 256, 0, stream>>>(x0, dinv, agg, N);
    k_edge<<<(int)((et + 255) / 256), 256, 0, stream>>>(ei, dinv, x0, agg, E);
    k_gemm32<<<gblocks, 128, 0, stream>>>(agg, W_c1, b_c1, x1, N);

    // conv2: x1 -> x2
    k_selfinit<<<(int)((nf + 255) / 256), 256, 0, stream>>>(x1, dinv, agg, N);
    k_edge<<<(int)((et + 255) / 256), 256, 0, stream>>>(ei, dinv, x1, agg, E);
    k_gemm32<<<gblocks, 128, 0, stream>>>(agg, W_c2, b_c2, x2, N);

    // conv3: x2 -> x3
    k_selfinit<<<(int)((nf + 255) / 256), 256, 0, stream>>>(x2, dinv, agg, N);
    k_edge<<<(int)((et + 255) / 256), 256, 0, stream>>>(ei, dinv, x2, agg, E);
    k_gemm32<<<gblocks, 128, 0, stream>>>(agg, W_c3, b_c3, x3, N);

    // concat + mynorm diffs + final linear
    k_final<<<tiles, 128, 0, stream>>>(x0, x1, x2, x3, W_fc4, b_fc4, out, N);
}
